// AttentionBlock_39367670235480
// MI455X (gfx1250) — matrix operand order repaired and checked by reading, not compiled, not run on hardware
//
#include <hip/hip_runtime.h>

// ---------------------------------------------------------------------------
// CDNA5 (gfx1250) attention block: GroupNorm -> QKV GEMM -> flash attention
// -> proj GEMM + residual.  GEMMs via v_wmma_f32_16x16x32_bf16 (wave32);
// attention K/V tiles staged block-wide through LDS with async loads
// (global_load_async_to_lds_b128 / ASYNCcnt), double buffered.
// ---------------------------------------------------------------------------

typedef __attribute__((ext_vector_type(16))) __bf16       v16bf;
typedef __attribute__((ext_vector_type(8)))  float        v8f;
typedef __attribute__((ext_vector_type(4)))  unsigned int u32x4;

struct Frag32B { u32x4 lo, hi; };

#define DEVINL __device__ __forceinline__

DEVINL v8f wmma_bf16(v16bf a, v16bf b, v8f c) {
    // D = A(16x32 bf16) * B(32x16 bf16) + C(16x16 f32)
    return __builtin_amdgcn_wmma_f32_16x16x32_bf16(
        /*neg_a=*/false, a, /*neg_b=*/false, b,
        /*c_mod=*/(short)0, c, /*reuse_a=*/false, /*reuse_b=*/false);
}

// A-matrix fragment (16x32, M x K), source row-major M x K with leading dim ld.
// ISA layout: lanes 0-15 row M=lane, K {k0..k0+7, k0+16..k0+23};
//             lanes 16-31 row M=lane-16, K {k0+8..15, k0+24..31}.
DEVINL v16bf load_afrag(const __bf16* base, int ld, int row0, int k0, int lane) {
    int r = lane & 15, hf = lane >> 4;
    const __bf16* p = base + (long)(row0 + r) * ld + k0 + hf * 8;
    Frag32B f;
    f.lo = *reinterpret_cast<const u32x4*>(p);
    f.hi = *reinterpret_cast<const u32x4*>(p + 16);
    return __builtin_bit_cast(v16bf, f);
}

// B-matrix fragment (32x16, K x N) where element(k,n) = W[n][k], W row-major
// N x K with leading dim ld (K contiguous).
// ISA layout: lane holds N=lane&15; lanes 0-15 K {k0..k0+7, k0+16..k0+23}; lanes 16-31 K {k0+8..15, k0+24..31}.
DEVINL v16bf load_bfrag(const __bf16* base, int ld, int n0, int k0, int lane) {
    int n = lane & 15, hf = lane >> 4;
    const __bf16* p = base + (long)(n0 + n) * ld + k0 + hf * 8;
    Frag32B f;
    f.lo = *reinterpret_cast<const u32x4*>(p);
    f.hi = *reinterpret_cast<const u32x4*>(p + 16);
    return __builtin_bit_cast(v16bf, f);
}

// ---------------------------------------------------------------------------
// Kernel 1: GroupNorm over (b, c, T=4096), 32 groups of 16 channels.
// Writes h transposed to (b, t, c) in bf16 so GEMM A-frags are contiguous.
// ---------------------------------------------------------------------------
__global__ void gn_kernel(const float* __restrict__ x,
                          const float* __restrict__ gamma,
                          const float* __restrict__ beta,
                          __bf16* __restrict__ h_t) {
    int bg = blockIdx.x;
    int b  = bg >> 5, g = bg & 31;
    int tid = threadIdx.x;
    __shared__ float s_sum[256], s_sq[256];

    const float* xp = x + ((long)(b * 512 + g * 16) << 12);
    float sum = 0.f, sq = 0.f;
    for (int i = tid; i < 16 * 4096; i += 256) {
        float v = xp[i];
        sum += v; sq += v * v;
    }
    s_sum[tid] = sum; s_sq[tid] = sq;
    __syncthreads();
    for (int s = 128; s > 0; s >>= 1) {
        if (tid < s) { s_sum[tid] += s_sum[tid + s]; s_sq[tid] += s_sq[tid + s]; }
        __syncthreads();
    }
    float mean = s_sum[0] * (1.0f / 65536.0f);
    float var  = s_sq[0] * (1.0f / 65536.0f) - mean * mean;
    float rstd = rsqrtf(var + 1e-5f);

    for (int i = tid; i < 16 * 4096; i += 256) {
        int cl = i >> 12, t = i & 4095;
        int c  = g * 16 + cl;
        float v = (xp[i] - mean) * rstd * gamma[c] + beta[c];
        h_t[((long)(b << 12) + t) * 512 + c] = (__bf16)v;
    }
}

// ---------------------------------------------------------------------------
// Kernel 2: convert GEMM weights to bf16.
// ---------------------------------------------------------------------------
__global__ void cvt_kernel(const float* __restrict__ wq, const float* __restrict__ wp,
                           __bf16* __restrict__ wqb, __bf16* __restrict__ wpb) {
    long i = (long)blockIdx.x * 256 + threadIdx.x;
    if (i < 1536 * 512) wqb[i] = (__bf16)wq[i];
    if (i < 512 * 512)  wpb[i] = (__bf16)wp[i];
}

// ---------------------------------------------------------------------------
// Kernel 3: QKV GEMM.  C[t,o] = sum_c h_t[b][t][c] * w_qkv[o][c] + b_qkv[o].
// Each wave: 16(t) x 64(o) tile, K=512 in 32-steps.
// q,k stored (bh, t, 64); v stored (bh, 64, t) -- all bf16.
// ---------------------------------------------------------------------------
__global__ void qkv_kernel(const __bf16* __restrict__ h_t,
                           const __bf16* __restrict__ wq,
                           const float*  __restrict__ bq,
                           __bf16* __restrict__ qb,
                           __bf16* __restrict__ kb,
                           __bf16* __restrict__ vb) {
    int lane = threadIdx.x & 31;
    int wid  = blockIdx.x * 8 + (threadIdx.x >> 5);
    int b    = wid / 6144;
    int rem  = wid % 6144;
    int mt   = rem / 24, ng = rem % 24;
    int t0   = mt * 16,  o0 = ng * 64;

    const __bf16* A = h_t + (long)b * 4096 * 512;
    const v8f vz = {0.f,0.f,0.f,0.f,0.f,0.f,0.f,0.f};
    v8f acc[4] = {vz, vz, vz, vz};

    for (int kc = 0; kc < 512; kc += 32) {
        if (kc + 32 < 512)
            __builtin_prefetch(A + (long)(t0 + (lane & 15)) * 512 + kc + 32, 0, 1);
        v16bf a = load_afrag(A, 512, t0, kc, lane);
#pragma unroll
        for (int j = 0; j < 4; ++j)
            acc[j] = wmma_bf16(a, load_bfrag(wq, 512, o0 + j * 16, kc, lane), acc[j]);
    }

    int n = lane & 15, hf = lane >> 4;
#pragma unroll
    for (int j = 0; j < 4; ++j) {
        int o    = o0 + j * 16 + n;
        int head = o / 192, oin = o % 192;
        float bias = bq[o];
#pragma unroll
        for (int r = 0; r < 8; ++r) {
            int t = t0 + r + hf * 8;
            __bf16 bv = (__bf16)(acc[j][r] + bias);
            if (oin < 64)
                qb[((long)(b * 8 + head) * 4096 + t) * 64 + oin] = bv;
            else if (oin < 128)
                kb[((long)(b * 8 + head) * 4096 + t) * 64 + (oin - 64)] = bv;
            else
                vb[((long)(b * 8 + head) * 64 + (oin - 128)) * 4096 + t] = bv;
        }
    }
}

// ---------------------------------------------------------------------------
// Kernel 4: flash attention.  All 8 waves of a block share one (b,head);
// each wave owns a 16-row query tile.  Per 32-wide key step the K(32x64) and
// V(64x32) tiles are staged block-wide into LDS with async loads (ASYNCcnt),
// double buffered so the DMA of step i+1 overlaps compute of step i.
// Online softmax in base-2 (v_exp_f32 is natively exp2).
// ---------------------------------------------------------------------------
__global__ void attn_kernel(const __bf16* __restrict__ qb,
                            const __bf16* __restrict__ kb,
                            const __bf16* __restrict__ vb,
                            __bf16* __restrict__ attn_t) {
    __shared__ __align__(16) __bf16 kt[2][32][64];    // [buf][s_local][c]  8KB
    __shared__ __align__(16) __bf16 vt[2][64][32];    // [buf][c][s_local]  8KB
    __shared__ __align__(16) __bf16 plds[8][16][32];  // per-wave P tile    8KB

    int tid  = threadIdx.x;
    int lane = tid & 31;
    int w    = tid >> 5;
    int wid  = blockIdx.x * 8 + w;
    int bh   = wid >> 8;                 // same for all 8 waves in the block
    int t0   = (wid & 255) << 4;

    const __bf16* qp = qb + (long)bh * 4096 * 64;
    const __bf16* kp = kb + (long)bh * 4096 * 64;
    const __bf16* vp = vb + (long)bh * 64 * 4096;

    v16bf aq0 = load_afrag(qp, 64, t0, 0,  lane);
    v16bf aq1 = load_afrag(qp, 64, t0, 32, lane);

    const v8f vz = {0.f,0.f,0.f,0.f,0.f,0.f,0.f,0.f};
    v8f ao[4] = {vz, vz, vz, vz};
    float mrun[8], lrun[8];
#pragma unroll
    for (int r = 0; r < 8; ++r) { mrun[r] = -1e30f; lrun[r] = 0.f; }

    int nn = lane & 15, hf = lane >> 4;

    // block-cooperative async stage of one 32-wide K/V step into buffer `buf`
    // K tile: 32 rows x 128B; V tile: 64 rows x 64B; 256 threads x 16B each.
    auto stage = [&](int s_next, int buf) {
        int krow = tid >> 3, kcol = (tid & 7) * 8;
        int vrow = tid >> 2, vcol = (tid & 3) * 8;
        const __bf16* kg = kp + (long)(s_next + krow) * 64 + kcol;
        const __bf16* vg = vp + (long)vrow * 4096 + s_next + vcol;
        unsigned klds = (unsigned)(size_t)&kt[buf][krow][kcol];
        unsigned vlds = (unsigned)(size_t)&vt[buf][vrow][vcol];
        asm volatile(
            "global_load_async_to_lds_b128 %0, %2, off\n\t"
            "global_load_async_to_lds_b128 %1, %3, off"
            :: "v"(klds), "v"(vlds), "v"(kg), "v"(vg)
            : "memory");
    };

    stage(0, 0);  // prologue: fill buffer 0

    const float SC = 0.125f * 1.44269504088896340736f;  // qk scale * log2(e)

    for (int s0 = 0; s0 < 4096; s0 += 32) {
        int buf = (s0 >> 5) & 1;
        // my async loads for `buf` done; barrier => everyone's done AND the
        // other buffer is free for the next stage.
        asm volatile("s_wait_asynccnt 0x0" ::: "memory");
        __syncthreads();
        if (s0 + 32 < 4096) stage(s0 + 32, buf ^ 1);

        const __bf16* kl = &kt[buf][0][0];
        const __bf16* vl = &vt[buf][0][0];

        // S = (Q*scale) . K^T over ch=64 (log2-domain scale folded in below)
        v8f s0a = vz, s1a = vz;
        s0a = wmma_bf16(aq0, load_bfrag(kl, 64, 0,  0,  lane), s0a);
        s0a = wmma_bf16(aq1, load_bfrag(kl, 64, 0,  32, lane), s0a);
        s1a = wmma_bf16(aq0, load_bfrag(kl, 64, 16, 0,  lane), s1a);
        s1a = wmma_bf16(aq1, load_bfrag(kl, 64, 16, 32, lane), s1a);

        float alpha[8];
#pragma unroll
        for (int r = 0; r < 8; ++r) {
            float v0 = s0a[r] * SC, v1 = s1a[r] * SC;
            // row lives in one half-wave (16 lanes): width-16 xor reduction
            float rmax = fmaxf(v0, v1);
#pragma unroll
            for (int m = 8; m >= 1; m >>= 1) rmax = fmaxf(rmax, __shfl_xor(rmax, m, 16));
            float mnew = fmaxf(mrun[r], rmax);
            float p0 = exp2f(v0 - mnew), p1 = exp2f(v1 - mnew);
            float psum = p0 + p1;
#pragma unroll
            for (int m = 8; m >= 1; m >>= 1) psum += __shfl_xor(psum, m, 16);
            alpha[r] = exp2f(mrun[r] - mnew);
            lrun[r]  = lrun[r] * alpha[r] + psum;
            mrun[r]  = mnew;
            int mrow = r + hf * 8;
            plds[w][mrow][nn]      = (__bf16)p0;   // cols s0..s0+15
            plds[w][mrow][nn + 16] = (__bf16)p1;   // cols s0+16..s0+31
        }
        asm volatile("s_wait_dscnt 0x0" ::: "memory");

        // Re-read P (16x32) in A-fragment layout from LDS (the D->A transpose).
        v16bf pa;
        {
            const __bf16* pb = &plds[w][0][0] + (long)(lane & 15) * 32 + hf * 8;
            Frag32B f;
            f.lo = *reinterpret_cast<const u32x4*>(pb);
            f.hi = *reinterpret_cast<const u32x4*>(pb + 16);
            pa = __builtin_bit_cast(v16bf, f);
        }

        // A_out(t, c) += P(t, s) * V(s, c);  V tile stored (c, s_local).
#pragma unroll
        for (int j = 0; j < 4; ++j) {
            v16bf bv = load_bfrag(vl, 32, j * 16, 0, lane);
#pragma unroll
            for (int r = 0; r < 8; ++r) ao[j][r] *= alpha[r];
            ao[j] = wmma_bf16(pa, bv, ao[j]);
        }
    }

    int b = bh >> 3, head = bh & 7;
#pragma unroll
    for (int r = 0; r < 8; ++r) {
        float inv = 1.0f / lrun[r];
        int t = t0 + r + hf * 8;
#pragma unroll
        for (int j = 0; j < 4; ++j) {
            int c = head * 64 + j * 16 + nn;
            attn_t[((long)(b * 4096 + t)) * 512 + c] = (__bf16)(ao[j][r] * inv);
        }
    }
}

// ---------------------------------------------------------------------------
// Kernel 5: out[b][o][t] = x[b][o][t] + sum_c w_proj[o][c]*attn[b][t][c] + b_proj[o]
// ---------------------------------------------------------------------------
__global__ void proj_kernel(const __bf16* __restrict__ attn_t,
                            const __bf16* __restrict__ wp,
                            const float*  __restrict__ bp,
                            const float*  __restrict__ x,
                            float* __restrict__ out) {
    int lane = threadIdx.x & 31;
    int wid  = blockIdx.x * 8 + (threadIdx.x >> 5);
    int b    = wid >> 11;
    int rem  = wid & 2047;
    int mt   = rem >> 3, ng = rem & 7;
    int t0   = mt * 16,  o0 = ng * 64;

    const __bf16* A = attn_t + (long)b * 4096 * 512;
    const v8f vz = {0.f,0.f,0.f,0.f,0.f,0.f,0.f,0.f};
    v8f acc[4] = {vz, vz, vz, vz};

    for (int kc = 0; kc < 512; kc += 32) {
        v16bf a = load_afrag(A, 512, t0, kc, lane);
#pragma unroll
        for (int j = 0; j < 4; ++j)
            acc[j] = wmma_bf16(a, load_bfrag(wp, 512, o0 + j * 16, kc, lane), acc[j]);
    }

    int n = lane & 15, hf = lane >> 4;
#pragma unroll
    for (int j = 0; j < 4; ++j) {
        int o = o0 + j * 16 + n;
        float bias = bp[o];
#pragma unroll
        for (int r = 0; r < 8; ++r) {
            int t = t0 + r + hf * 8;
            long idx = ((long)(b * 512 + o) << 12) + t;
            out[idx] = x[idx] + acc[j][r] + bias;
        }
    }
}

// ---------------------------------------------------------------------------
// Launch
// ---------------------------------------------------------------------------
extern "C" void kernel_launch(void* const* d_in, const int* in_sizes, int n_in,
                              void* d_out, int out_size, void* d_ws, size_t ws_size,
                              hipStream_t stream) {
    const float* x     = (const float*)d_in[0];
    const float* gamma = (const float*)d_in[1];
    const float* beta  = (const float*)d_in[2];
    const float* w_qkv = (const float*)d_in[3];
    const float* b_qkv = (const float*)d_in[4];
    const float* w_prj = (const float*)d_in[5];
    const float* b_prj = (const float*)d_in[6];
    float* out = (float*)d_out;

    char* ws = (char*)d_ws;
    // workspace layout (bytes)
    __bf16* h_t    = (__bf16*)(ws + 0);           //  8,388,608  (2,4096,512) bf16
    __bf16* wqkv_b = (__bf16*)(ws + 8388608);     //  1,572,864  (1536,512)
    __bf16* wprj_b = (__bf16*)(ws + 9961472);     //    524,288  (512,512)
    __bf16* qbuf   = (__bf16*)(ws + 10485760);    //  8,388,608  (16,4096,64)
    __bf16* kbuf   = (__bf16*)(ws + 18874368);    //  8,388,608  (16,4096,64)
    __bf16* vbuf   = (__bf16*)(ws + 27262976);    //  8,388,608  (16,64,4096)
    __bf16* attn_t = (__bf16*)(ws + 35651584);    //  8,388,608  (2,4096,512)

    gn_kernel  <<<64,   256, 0, stream>>>(x, gamma, beta, h_t);
    cvt_kernel <<<3072, 256, 0, stream>>>(w_qkv, w_prj, wqkv_b, wprj_b);
    qkv_kernel <<<1536, 256, 0, stream>>>(h_t, wqkv_b, b_qkv, qbuf, kbuf, vbuf);
    attn_kernel<<<512,  256, 0, stream>>>(qbuf, kbuf, vbuf, attn_t);
    proj_kernel<<<512,  256, 0, stream>>>(attn_t, wprj_b, b_prj, x, out);
}
